// MultiHeadAttention_44994077393098
// MI455X (gfx1250) — matrix operand order repaired and checked by reading, not compiled, not run on hardware
//
#include <hip/hip_runtime.h>
#include <hip/hip_bf16.h>

constexpr int C_B = 4, C_T = 4096, C_D = 512, C_H = 8, C_DH = 64;
constexpr int C_BLK = 128, C_NB = C_T / C_BLK;

typedef __attribute__((ext_vector_type(16))) __bf16 v16bf;
typedef __attribute__((ext_vector_type(8)))  __bf16 v8bf;
typedef __attribute__((ext_vector_type(8)))  float  v8f;

// A-matrix 16x32 bf16 fragment: lane half g reads row chunks [g*8,g*8+8) and [16+g*8,+8)
static __device__ __forceinline__ v16bf afrag_bf16(const __bf16* p, int g) {
    v16bf f;
    v8bf a = *reinterpret_cast<const v8bf*>(p + g * 8);
    v8bf b = *reinterpret_cast<const v8bf*>(p + 16 + g * 8);
#pragma unroll
    for (int i = 0; i < 8; ++i) { f[i] = a[i]; f[i + 8] = b[i]; }
    return f;
}
static __device__ __forceinline__ v16bf afrag_f32(const float* p, int g) {
    v16bf f;
#pragma unroll
    for (int i = 0; i < 8; ++i) {
        f[i]     = (__bf16)p[g * 8 + i];
        f[i + 8] = (__bf16)p[16 + g * 8 + i];
    }
    return f;
}
// B-matrix 32x16 bf16 fragment: lane half g reads column chunks [g*8,g*8+8) and [16+g*8,+8)
static __device__ __forceinline__ v16bf bfrag_bf16(const __bf16* p, int g) {
    v16bf f;
    v8bf a = *reinterpret_cast<const v8bf*>(p + g * 8);
    v8bf b = *reinterpret_cast<const v8bf*>(p + 16 + g * 8);
#pragma unroll
    for (int i = 0; i < 8; ++i) { f[i] = a[i]; f[i + 8] = b[i]; }
    return f;
}
static __device__ __forceinline__ v8f wmma_bf16(v16bf a, v16bf b, v8f c) {
    return __builtin_amdgcn_wmma_f32_16x16x32_bf16(false, a, false, b, (short)0, c,
                                                   false, false);
}

// ---------------------------------------------------------------------------
// Kernel 1: qkv = x @ W_qkv (M=16384, N=1536, K=512), fused RoPE epilogue.
// Grid (12, 128); block 256 = 8 waves, each wave owns a 32x64 output sub-tile
// (so each wave covers exactly one head's 64 columns -> RoPE pairing local).
// ---------------------------------------------------------------------------
__global__ __launch_bounds__(256) void qkv_rope_kernel(
    const float* __restrict__ x, const float* __restrict__ Wqkv,
    __bf16* __restrict__ qo, __bf16* __restrict__ ko, __bf16* __restrict__ vo)
{
    __shared__ __bf16 ldsB[128 * 32];  // W tile, transposed [n][k]
    const int tid = threadIdx.x;
    const int lane = tid & 31, w = tid >> 5;
    const int lanen = lane & 15, hg = lane >> 4;
    const int ncb = blockIdx.x * 128;   // column tile base in [0,1536)
    const int mrb = blockIdx.y * 128;   // row tile base in [0,16384)
    const int wm = w & 3, wn = w >> 2;  // wave rows = wm*32, wave cols = wn*64

    v8f z = {0.f, 0.f, 0.f, 0.f, 0.f, 0.f, 0.f, 0.f};
    v8f acc[2][4];
#pragma unroll
    for (int rt = 0; rt < 2; ++rt)
#pragma unroll
        for (int nt = 0; nt < 4; ++nt) acc[rt][nt] = z;

    for (int ks = 0; ks < 16; ++ks) {
        const int kb = ks * 32;
#pragma unroll
        for (int j = 0; j < 16; ++j) {
            int lin = j * 256 + tid;
            int kk = lin >> 7, nn = lin & 127;
            ldsB[nn * 32 + kk] = (__bf16)Wqkv[(size_t)(kb + kk) * 1536 + ncb + nn];
        }
        __syncthreads();
        v16bf bf[4];
#pragma unroll
        for (int nt = 0; nt < 4; ++nt)
            bf[nt] = bfrag_bf16(&ldsB[(wn * 64 + nt * 16 + lanen) * 32], hg);
#pragma unroll
        for (int rt = 0; rt < 2; ++rt) {
            const float* ap = x + (size_t)(mrb + wm * 32 + rt * 16 + lanen) * C_D + kb;
            v16bf af = afrag_f32(ap, hg);
#pragma unroll
            for (int nt = 0; nt < 4; ++nt)
                acc[rt][nt] = wmma_bf16(af, bf[nt], acc[rt][nt]);
        }
        __syncthreads();
    }

    const int gcb = ncb + wn * 64;       // global column base (aligned to 64)
    const int sec = gcb / C_D;           // 0=q 1=k 2=v
    const int head = (gcb % C_D) / C_DH;
#pragma unroll
    for (int rt = 0; rt < 2; ++rt) {
#pragma unroll
        for (int r = 0; r < 8; ++r) {
            int row = mrb + wm * 32 + rt * 16 + r + 8 * hg;  // = b*T + t
            int bb = row >> 12, tt = row & (C_T - 1);
            size_t base = ((size_t)(bb * C_H + head) * C_T + tt) * C_DH;
            if (sec == 2) {
#pragma unroll
                for (int nt = 0; nt < 4; ++nt)
                    vo[base + nt * 16 + lanen] = (__bf16)acc[rt][nt][r];
            } else {
                __bf16* dst = (sec == 0) ? qo : ko;
#pragma unroll
                for (int nt = 0; nt < 2; ++nt) {
                    int dh1 = nt * 16 + lanen;  // < 32
                    float x1 = acc[rt][nt][r];
                    float x2 = acc[rt][nt + 2][r];
                    // inv_freq = 10000^(-2*dh1/64) = exp(-0.287823...*dh1)
                    float ang = (float)tt * __expf(-0.2878231366242557f * (float)dh1);
                    float sn, cs;
                    __sincosf(ang, &sn, &cs);
                    dst[base + dh1]      = (__bf16)(x1 * cs - x2 * sn);
                    dst[base + dh1 + 32] = (__bf16)(x1 * sn + x2 * cs);
                }
            }
        }
    }
}

// ---------------------------------------------------------------------------
// Kernel 2: block-local attention with online softmax.
// Grid (32, 8, 4) = (qblock, head, batch); 8 waves, wave w owns query rows
// [w*16, w*16+16). Loops over 3 key blocks of 128 (edge blocks skipped).
// ---------------------------------------------------------------------------
__global__ __launch_bounds__(256) void attn_kernel(
    const __bf16* __restrict__ q, const __bf16* __restrict__ k,
    const __bf16* __restrict__ v, __bf16* __restrict__ ob)
{
    __shared__ __bf16 ldsP[128 * 136];   // probabilities [qrow][key]
    __shared__ __bf16 ldsVT[64 * 136];   // V transposed [dh][key]
    const int tid = threadIdx.x;
    const int lane = tid & 31, w = tid >> 5;
    const int lanen = lane & 15, hg = lane >> 4;
    const int qb = blockIdx.x, h = blockIdx.y, b = blockIdx.z;
    const size_t bh = (size_t)(b * C_H + h) * C_T;
    const int wrow = w * 16;

    // Q A-fragments (rows fixed per wave), straight from global bf16
    v16bf aq[2];
#pragma unroll
    for (int ksp = 0; ksp < 2; ++ksp)
        aq[ksp] = afrag_bf16(q + (bh + qb * C_BLK + wrow + lanen) * C_DH + ksp * 32, hg);

    v8f z = {0.f, 0.f, 0.f, 0.f, 0.f, 0.f, 0.f, 0.f};
    v8f oacc[4];
#pragma unroll
    for (int nt = 0; nt < 4; ++nt) oacc[nt] = z;
    float m[8], l[8];
#pragma unroll
    for (int r = 0; r < 8; ++r) { m[r] = -1.0e4f; l[r] = 0.f; }

    for (int kb = 0; kb < 3; ++kb) {
        if (kb == 0 && qb == 0) continue;          // keys entirely before t=0
        if (kb == 2 && qb == C_NB - 1) continue;   // keys entirely past t=T
        const int gk0 = (qb + kb - 1) * C_BLK;     // first key token (in range)

        // stage V^T in LDS
#pragma unroll
        for (int j = 0; j < 32; ++j) {
            int lin = j * 256 + tid;
            int tok = lin >> 6, dh = lin & 63;
            ldsVT[dh * 136 + tok] = v[(bh + gk0 + tok) * C_DH + dh];
        }

        // S = q @ k^T  (128x128 per workgroup; 16x128 per wave)
        v8f sacc[8];
#pragma unroll
        for (int nt = 0; nt < 8; ++nt) sacc[nt] = z;
#pragma unroll
        for (int nt = 0; nt < 8; ++nt) {
            const __bf16* kp = k + (bh + gk0 + nt * 16 + lanen) * C_DH;
#pragma unroll
            for (int ksp = 0; ksp < 2; ++ksp) {
                v16bf bk = bfrag_bf16(kp + ksp * 32, hg);
                sacc[nt] = wmma_bf16(aq[ksp], bk, sacc[nt]);
            }
        }

        // scale + band mask: allowed iff 1 <= kj - qi <= 256
#pragma unroll
        for (int nt = 0; nt < 8; ++nt) {
            int kj = kb * 128 + nt * 16 + lanen;
#pragma unroll
            for (int r = 0; r < 8; ++r) {
                int qi = wrow + r + 8 * hg;
                int d = kj - qi;
                float s = sacc[nt][r] * 0.125f;
                sacc[nt][r] = (d >= 1 && d <= 256) ? s : -1.0e30f;
            }
        }

        // online softmax: row lives across a 16-lane half + 8 tiles
#pragma unroll
        for (int r = 0; r < 8; ++r) {
            float pm = -1.0e30f;
#pragma unroll
            for (int nt = 0; nt < 8; ++nt) pm = fmaxf(pm, sacc[nt][r]);
            pm = fmaxf(pm, __shfl_xor(pm, 1, 32));
            pm = fmaxf(pm, __shfl_xor(pm, 2, 32));
            pm = fmaxf(pm, __shfl_xor(pm, 4, 32));
            pm = fmaxf(pm, __shfl_xor(pm, 8, 32));
            float mn = fmaxf(m[r], pm);        // m floored at -1e4: masked exp -> 0
            float alpha = __expf(m[r] - mn);
            m[r] = mn;
            float ps = 0.f;
#pragma unroll
            for (int nt = 0; nt < 8; ++nt) {
                float p = __expf(sacc[nt][r] - mn);
                sacc[nt][r] = p;
                ps += p;
            }
            ps += __shfl_xor(ps, 1, 32);
            ps += __shfl_xor(ps, 2, 32);
            ps += __shfl_xor(ps, 4, 32);
            ps += __shfl_xor(ps, 8, 32);
            l[r] = l[r] * alpha + ps;
#pragma unroll
            for (int nt = 0; nt < 4; ++nt) oacc[nt][r] *= alpha;
        }

        // route P through LDS to re-enter A layout
#pragma unroll
        for (int nt = 0; nt < 8; ++nt)
#pragma unroll
            for (int r = 0; r < 8; ++r)
                ldsP[(wrow + r + 8 * hg) * 136 + nt * 16 + lanen] = (__bf16)sacc[nt][r];
        __syncthreads();

        // O += P @ V
#pragma unroll
        for (int ksp = 0; ksp < 4; ++ksp) {
            v16bf ap = afrag_bf16(&ldsP[(wrow + lanen) * 136 + ksp * 32], hg);
#pragma unroll
            for (int nt = 0; nt < 4; ++nt) {
                v16bf bv = bfrag_bf16(&ldsVT[(nt * 16 + lanen) * 136 + ksp * 32], hg);
                oacc[nt] = wmma_bf16(ap, bv, oacc[nt]);
            }
        }
        __syncthreads();
    }

    // normalize and write ob[b][t][h][dh] (== [B,T,D] contiguous)
#pragma unroll
    for (int nt = 0; nt < 4; ++nt)
#pragma unroll
        for (int r = 0; r < 8; ++r) {
            int tok = qb * C_BLK + wrow + r + 8 * hg;
            float val = oacc[nt][r] / l[r];
            ob[((size_t)b * C_T + tok) * C_D + h * C_DH + nt * 16 + lanen] = (__bf16)val;
        }
}

// ---------------------------------------------------------------------------
// Kernel 3: out = ob @ W_out + b_out (M=16384, N=512, K=512), fp32 output.
// ---------------------------------------------------------------------------
__global__ __launch_bounds__(256) void outproj_kernel(
    const __bf16* __restrict__ ob, const float* __restrict__ Wout,
    const float* __restrict__ bout, float* __restrict__ out)
{
    __shared__ __bf16 ldsB[128 * 32];
    const int tid = threadIdx.x;
    const int lane = tid & 31, w = tid >> 5;
    const int lanen = lane & 15, hg = lane >> 4;
    const int ncb = blockIdx.x * 128;
    const int mrb = blockIdx.y * 128;
    const int wm = w & 3, wn = w >> 2;

    v8f z = {0.f, 0.f, 0.f, 0.f, 0.f, 0.f, 0.f, 0.f};
    v8f acc[2][4];
#pragma unroll
    for (int rt = 0; rt < 2; ++rt)
#pragma unroll
        for (int nt = 0; nt < 4; ++nt) acc[rt][nt] = z;

    for (int ks = 0; ks < 16; ++ks) {
        const int kb = ks * 32;
#pragma unroll
        for (int j = 0; j < 16; ++j) {
            int lin = j * 256 + tid;
            int kk = lin >> 7, nn = lin & 127;
            ldsB[nn * 32 + kk] = (__bf16)Wout[(size_t)(kb + kk) * C_D + ncb + nn];
        }
        __syncthreads();
        v16bf bf[4];
#pragma unroll
        for (int nt = 0; nt < 4; ++nt)
            bf[nt] = bfrag_bf16(&ldsB[(wn * 64 + nt * 16 + lanen) * 32], hg);
#pragma unroll
        for (int rt = 0; rt < 2; ++rt) {
            v16bf af = afrag_bf16(
                ob + (size_t)(mrb + wm * 32 + rt * 16 + lanen) * C_D + kb, hg);
#pragma unroll
            for (int nt = 0; nt < 4; ++nt)
                acc[rt][nt] = wmma_bf16(af, bf[nt], acc[rt][nt]);
        }
        __syncthreads();
    }

#pragma unroll
    for (int rt = 0; rt < 2; ++rt)
#pragma unroll
        for (int nt = 0; nt < 4; ++nt) {
            int col = ncb + wn * 64 + nt * 16 + lanen;
            float bias = bout[col];
#pragma unroll
            for (int r = 0; r < 8; ++r) {
                int row = mrb + wm * 32 + rt * 16 + r + 8 * hg;
                out[(size_t)row * C_D + col] = acc[rt][nt][r] + bias;
            }
        }
}

extern "C" void kernel_launch(void* const* d_in, const int* in_sizes, int n_in,
                              void* d_out, int out_size, void* d_ws, size_t ws_size,
                              hipStream_t stream)
{
    const float* x    = (const float*)d_in[0];
    const float* Wqkv = (const float*)d_in[1];
    const float* Wout = (const float*)d_in[2];
    const float* bout = (const float*)d_in[3];
    float* out = (float*)d_out;

    char* ws = (char*)d_ws;
    const size_t NE = (size_t)C_B * C_H * C_T * C_DH;  // 8,388,608 elements
    __bf16* q  = (__bf16*)(ws);
    __bf16* k  = (__bf16*)(ws + NE * 2);
    __bf16* v  = (__bf16*)(ws + NE * 4);
    __bf16* ob = (__bf16*)(ws + NE * 6);

    qkv_rope_kernel<<<dim3(12, 128), 256, 0, stream>>>(x, Wqkv, q, k, v);
    attn_kernel<<<dim3(C_NB, C_H, C_B), 256, 0, stream>>>(q, k, v, ob);
    outproj_kernel<<<dim3(4, 128), 256, 0, stream>>>(ob, Wout, bout, out);

    (void)in_sizes; (void)n_in; (void)out_size; (void)ws_size;
}
